// PNA_40733469835825
// MI455X (gfx1250) — compile-verified, matrix operand order checked by reading, not run on hardware
//
#include <hip/hip_runtime.h>
#include <hip/hip_bf16.h>
#include <cmath>

// ---------------- problem constants ----------------
#define NN 10000      // nodes
#define NE 100000     // edges
#define NG 64         // graphs
#define FT 75         // hidden features
#define TW 5          // towers
#define FO 15         // tower output features
#define NL 4          // layers
#define KPAD 256      // 225 padded to 8 chunks of 32 (f16 WMMA K)
#define NTILES 25     // 5 towers * ceil(75/16)=5 N-tiles
#define MSTRIDE 384   // padded row stride of m16 (5*75=375 -> 384)

typedef __attribute__((ext_vector_type(16))) _Float16 v16h;
typedef __attribute__((ext_vector_type(8)))  _Float16 v8h;
typedef __attribute__((ext_vector_type(8)))  float    v8f;

static inline int cdiv(long long a, long long b) { return (int)((a + b - 1) / b); }

// ---------------- generic fills ----------------
__global__ void k_fill(float* __restrict__ p, float v, int n) {
  int i = blockIdx.x * blockDim.x + threadIdx.x;
  if (i < n) p[i] = v;
}
__global__ void k_filli(int* __restrict__ p, int v, int n) {
  int i = blockIdx.x * blockDim.x + threadIdx.x;
  if (i < n) p[i] = v;
}

// ---------------- embeddings ----------------
__global__ void k_node_emb(const float* __restrict__ x, const float* __restrict__ w,
                           const float* __restrict__ b, float* __restrict__ h) {
  int i = blockIdx.x * blockDim.x + threadIdx.x;
  if (i >= NN * FT) return;
  int n = i / FT, f = i % FT;
  float acc = b[f];
  #pragma unroll
  for (int k = 0; k < 14; ++k) acc += x[n * 14 + k] * w[k * FT + f];
  h[i] = acc;
}

__global__ void k_edge_emb(const float* __restrict__ ea, const float* __restrict__ w,
                           const float* __restrict__ b, float* __restrict__ out) {
  int i = blockIdx.x * blockDim.x + threadIdx.x;
  if (i >= NE * 50) return;
  int e = i / 50, j = i % 50;
  float acc = b[j];
  #pragma unroll
  for (int k = 0; k < 4; ++k) acc += ea[e * 4 + k] * w[k * 50 + j];
  out[i] = acc;
}

// ---------------- in-degree + CSR build (dst is launch-invariant) ----------------
__global__ void k_count(const int* __restrict__ dst, float* __restrict__ cnt) {
  int e = blockIdx.x * blockDim.x + threadIdx.x;
  if (e < NE) atomicAdd(&cnt[dst[e]], 1.0f);
}

// one-block exclusive scan of degrees -> rowstart
__global__ void k_scan(const float* __restrict__ cnt, int* __restrict__ rowstart) {
  __shared__ int part[256];
  int tid = threadIdx.x;
  const int chunk = (NN + 255) / 256;  // 40
  int base = tid * chunk;
  int s = 0;
  for (int i = 0; i < chunk; ++i) {
    int n = base + i;
    if (n < NN) s += (int)cnt[n];
  }
  part[tid] = s;
  __syncthreads();
  if (tid == 0) {
    int run = 0;
    for (int i = 0; i < 256; ++i) { int v = part[i]; part[i] = run; run += v; }
  }
  __syncthreads();
  int run = part[tid];
  for (int i = 0; i < chunk; ++i) {
    int n = base + i;
    if (n < NN) { rowstart[n] = run; run += (int)cnt[n]; }
  }
}

__global__ void k_scatter(const int* __restrict__ dst, const int* __restrict__ rowstart,
                          int* __restrict__ fill, int* __restrict__ eidx) {
  int e = blockIdx.x * blockDim.x + threadIdx.x;
  if (e >= NE) return;
  int d = dst[e];
  int pos = rowstart[d] + atomicAdd(&fill[d], 1);
  eidx[pos] = e;
}

// ---------------- WMMA B-fragment pre-shuffle of pre_w (all layers) ----------------
// Bw layout: [layer][tn][kc][lane][16 halfs]; tn = t*5 + ntile
__global__ void k_bwconv(const float* __restrict__ pre_w, _Float16* __restrict__ Bw) {
  int id = blockIdx.x * blockDim.x + threadIdx.x;   // (((layer*25+tn)*8+kc)*32+lane)
  if (id >= NL * NTILES * 8 * 32) return;
  int lane  = id & 31;
  int kc    = (id >> 5) & 7;
  int tn    = (id >> 8) % NTILES;
  int layer = id / (NTILES * 8 * 32);
  int t = tn / 5, nt = tn % 5;
  int n = nt * 16 + (lane & 15);
  int khalf = lane >> 4;
  _Float16* out = Bw + (size_t)id * 16;
  #pragma unroll
  for (int i = 0; i < 16; ++i) {
    int klocal = (i < 8) ? (khalf * 8 + i) : (khalf * 8 + i + 8);
    int k = kc * 32 + klocal;
    float v = 0.0f;
    if (k < 3 * FT && n < FT)
      v = pre_w[(((size_t)layer * TW + t) * (3 * FT) + k) * FT + n];
    out[i] = (_Float16)v;
  }
}

// ---------------- build fp16 A matrix [E][KPAD] = [h[dst] | h[src] | edge_enc] ------
// one thread per 8-half group; edge-encoder GEMM (50->75) fused; one b128 store
__global__ void k_stageA(const float* __restrict__ h, const float* __restrict__ eaemb,
                         const float* __restrict__ eew, const float* __restrict__ eeb,
                         const int* __restrict__ src, const int* __restrict__ dst,
                         _Float16* __restrict__ A) {
  long long id = (long long)blockIdx.x * blockDim.x + threadIdx.x;
  if (id >= (long long)NE * (KPAD / 8)) return;
  int grp = (int)(id & 31);
  int e = (int)(id >> 5);
  int c0 = grp * 8;
  int nd = dst[e], ns = src[e];
  v8h outv;
  #pragma unroll
  for (int i = 0; i < 8; ++i) {
    int c = c0 + i;
    float v = 0.0f;
    if (c < FT) {
      v = h[(size_t)nd * FT + c];
    } else if (c < 2 * FT) {
      v = h[(size_t)ns * FT + (c - FT)];
    } else if (c < 3 * FT) {
      int f = c - 2 * FT;
      float acc = eeb[f];
      for (int j = 0; j < 50; ++j) acc += eaemb[(size_t)e * 50 + j] * eew[j * FT + f];
      v = acc;
    }
    outv[i] = (_Float16)v;
  }
  *(v8h*)(A + (size_t)e * KPAD + c0) = outv;
}

// ---------------- hot kernel: per-tower pre-NN GEMM via WMMA, store m16 ----------
// one wave per 16-edge stripe; 25 N-tiles (5 towers x 5), K = 8 chunks of 32 f16
__global__ void __launch_bounds__(256)
k_wmma_pre(const _Float16* __restrict__ A, const _Float16* __restrict__ Bw,
           const float* __restrict__ pb, _Float16* __restrict__ m16) {
  int wave = (blockIdx.x * blockDim.x + threadIdx.x) >> 5;
  int lane = threadIdx.x & 31;
  if (wave * 16 >= NE) return;
  int m0 = wave * 16;
  int row = lane & 15;
  int khalf = lane >> 4;

  // preload all 8 A fragments for this stripe (held in VGPRs, reused 25x)
  const _Float16* abase = A + (size_t)(m0 + row) * KPAD + khalf * 8;
  v16h af[8];
  #pragma unroll
  for (int kc = 0; kc < 8; ++kc) {
    v8h lo = *(const v8h*)(abase + kc * 32);
    v8h hi = *(const v8h*)(abase + kc * 32 + 16);
    v16h a;
    #pragma unroll
    for (int i = 0; i < 8; ++i) { a[i] = lo[i]; a[i + 8] = hi[i]; }
    af[kc] = a;
  }

  for (int tn = 0; tn < NTILES; ++tn) {
    int t = tn / 5, nt = tn % 5;
    v8f acc = {};
    const _Float16* bbase = Bw + ((size_t)tn * 256 + lane) * 16;
    #pragma unroll
    for (int kc = 0; kc < 8; ++kc) {
      v8h blo = *(const v8h*)(bbase + (size_t)kc * 512);
      v8h bhi = *(const v8h*)(bbase + (size_t)kc * 512 + 8);
      v16h b;
      #pragma unroll
      for (int i = 0; i < 8; ++i) { b[i] = blo[i]; b[i + 8] = bhi[i]; }
      acc = __builtin_amdgcn_wmma_f32_16x16x32_f16(
          /*neg_a=*/false, af[kc], /*neg_b=*/false, b,
          /*c_mod=*/(short)0, acc, /*reuse_a=*/false, /*reuse_b=*/false);
    }
    // epilogue: plain fp16 stores of m (+bias) -- no atomics
    int n = nt * 16 + (lane & 15);
    if (n < FT) {
      float bias = pb[t * FT + n];
      #pragma unroll
      for (int r = 0; r < 8; ++r) {
        int ed = m0 + ((lane >> 4) ? (8 + r) : r);
        m16[(size_t)ed * MSTRIDE + t * FT + n] = (_Float16)(acc[r] + bias);
      }
    }
  }
}

// ---------------- fused CSR gather-aggregate + post-NN ----------------
// one block per node, 5 waves (one per tower); lanes read contiguous 64B rows of m16
__global__ void __launch_bounds__(160)
k_agg_post(const float* __restrict__ h, const _Float16* __restrict__ m16,
           const int* __restrict__ rowstart, const float* __restrict__ cnt,
           const int* __restrict__ eidx,
           const float* __restrict__ pw, const float* __restrict__ pb, float avg_log,
           float* __restrict__ out) {
  __shared__ float agg[TW][304];
  __shared__ float hL[80];
  __shared__ float sclL[2];
  int n = blockIdx.x;
  int tid = threadIdx.x;
  int t = tid >> 5;
  int lane = tid & 31;
  float c = cnt[n];
  float c1 = fmaxf(c, 1.0f);
  int start = rowstart[n];
  int deg = (int)c;
  for (int f = lane; f < FT; f += 32) {
    float s = 0.0f, s2 = 0.0f;
    float mn = 3.402823466e+38f, mx = -3.402823466e+38f;
    for (int j = 0; j < deg; ++j) {
      int e = eidx[start + j];
      float v = (float)m16[(size_t)e * MSTRIDE + t * FT + f];
      s += v; s2 += v * v;
      mn = fminf(mn, v); mx = fmaxf(mx, v);
    }
    float mean = s / c1;
    float var = s2 / c1 - mean * mean;
    float sd = sqrtf(fmaxf(var, 0.0f) + 1e-5f);
    if (deg == 0) { mn = 0.0f; mx = 0.0f; }
    agg[t][f] = mean; agg[t][FT + f] = mn; agg[t][2 * FT + f] = mx; agg[t][3 * FT + f] = sd;
  }
  if (t == 0) {
    for (int f = lane; f < FT; f += 32) hL[f] = h[(size_t)n * FT + f];
    if (lane == 0) {
      float s = logf(c1 + 1.0f) / avg_log;
      sclL[0] = s; sclL[1] = 1.0f / s;
    }
  }
  __syncthreads();
  if (lane < FO) {
    int g = lane;
    const float* W = pw + (size_t)t * (13 * FT) * FO;
    float acc = pb[t * FO + g];
    for (int k = 0; k < FT; ++k) acc += hL[k] * W[k * FO + g];
    float scl = sclL[0], inv = sclL[1];
    for (int r = 0; r < 4 * FT; ++r) {
      float a = agg[t][r];
      acc += a * (W[(FT + r) * FO + g] + scl * W[(5 * FT + r) * FO + g]
                                       + inv * W[(9 * FT + r) * FO + g]);
    }
    out[(size_t)n * FT + t * FO + g] = acc;
  }
}

// ---------------- lin: [N,75] @ [75,75] ----------------
__global__ void k_lin(const float* __restrict__ po, const float* __restrict__ w,
                      const float* __restrict__ b, float* __restrict__ out) {
  int i = blockIdx.x * blockDim.x + threadIdx.x;
  if (i >= NN * FT) return;
  int n = i / FT, f = i % FT;
  float acc = b[f];
  for (int k = 0; k < FT; ++k) acc += po[(size_t)n * FT + k] * w[k * FT + f];
  out[i] = acc;
}

// ---------------- batchnorm (biased var) ----------------
__global__ void k_bnstats(const float* __restrict__ v, float* __restrict__ mu,
                          float* __restrict__ rs) {
  __shared__ float s1[256], s2[256];
  int f = blockIdx.x, tid = threadIdx.x;
  float a = 0.0f, b = 0.0f;
  for (int n = tid; n < NN; n += 256) {
    float x = v[(size_t)n * FT + f];
    a += x; b += x * x;
  }
  s1[tid] = a; s2[tid] = b;
  __syncthreads();
  for (int s = 128; s > 0; s >>= 1) {
    if (tid < s) { s1[tid] += s1[tid + s]; s2[tid] += s2[tid + s]; }
    __syncthreads();
  }
  if (tid == 0) {
    float m = s1[0] / (float)NN;
    float var = s2[0] / (float)NN - m * m;
    mu[f] = m;
    rs[f] = rsqrtf(var + 1e-5f);
  }
}

__global__ void k_bnapply(const float* __restrict__ v, const float* __restrict__ mu,
                          const float* __restrict__ rs, const float* __restrict__ g,
                          const float* __restrict__ b, float* __restrict__ out) {
  int i = blockIdx.x * blockDim.x + threadIdx.x;
  if (i >= NN * FT) return;
  int f = i % FT;
  out[i] = fmaxf((v[i] - mu[f]) * rs[f] * g[f] + b[f], 0.0f);
}

// ---------------- pooling + head MLP ----------------
__global__ void k_copy(const float* __restrict__ h, float* __restrict__ out) {
  int i = blockIdx.x * blockDim.x + threadIdx.x;
  if (i < NN * FT) out[i] = h[i];
}

__global__ void k_pool(const float* __restrict__ h, const int* __restrict__ batch,
                       float* __restrict__ gp) {
  int i = blockIdx.x * blockDim.x + threadIdx.x;
  if (i >= NN * FT) return;
  int n = i / FT, f = i % FT;
  atomicAdd(&gp[(size_t)batch[n] * FT + f], h[i]);
}

__global__ void k_mlp(const float* __restrict__ gp,
                      const float* __restrict__ w1, const float* __restrict__ b1,
                      const float* __restrict__ w2, const float* __restrict__ b2,
                      const float* __restrict__ w3, const float* __restrict__ b3,
                      float* __restrict__ out) {
  __shared__ float hin[FT], h1[50], h2[25];
  int g = blockIdx.x, tid = threadIdx.x;
  for (int f = tid; f < FT; f += 64) hin[f] = gp[(size_t)g * FT + f];
  __syncthreads();
  if (tid < 50) {
    float a = b1[tid];
    for (int k = 0; k < FT; ++k) a += hin[k] * w1[k * 50 + tid];
    h1[tid] = fmaxf(a, 0.0f);
  }
  __syncthreads();
  if (tid < 25) {
    float a = b2[tid];
    for (int k = 0; k < 50; ++k) a += h1[k] * w2[k * 25 + tid];
    h2[tid] = fmaxf(a, 0.0f);
  }
  __syncthreads();
  if (tid == 0) {
    float a = b3[0];
    for (int k = 0; k < 25; ++k) a += h2[k] * w3[k];
    out[g] = a;
  }
}

// ---------------- host driver ----------------
extern "C" void kernel_launch(void* const* d_in, const int* in_sizes, int n_in,
                              void* d_out, int out_size, void* d_ws, size_t ws_size,
                              hipStream_t stream) {
  const float* x          = (const float*)d_in[0];
  const int*   ei         = (const int*)d_in[1];
  const float* edge_attr  = (const float*)d_in[2];
  const int*   batch      = (const int*)d_in[3];
  const float* node_emb_w = (const float*)d_in[4];
  const float* node_emb_b = (const float*)d_in[5];
  const float* edge_emb_w = (const float*)d_in[6];
  const float* edge_emb_b = (const float*)d_in[7];
  const float* ee_w       = (const float*)d_in[8];
  const float* ee_b       = (const float*)d_in[9];
  const float* pre_w      = (const float*)d_in[10];
  const float* pre_b      = (const float*)d_in[11];
  const float* post_w     = (const float*)d_in[12];
  const float* post_b     = (const float*)d_in[13];
  const float* lin_w      = (const float*)d_in[14];
  const float* lin_b      = (const float*)d_in[15];
  const float* bn_g       = (const float*)d_in[16];
  const float* bn_b       = (const float*)d_in[17];
  const float* mlp_w1     = (const float*)d_in[18];
  const float* mlp_b1     = (const float*)d_in[19];
  const float* mlp_w2     = (const float*)d_in[20];
  const float* mlp_b2     = (const float*)d_in[21];
  const float* mlp_w3     = (const float*)d_in[22];
  const float* mlp_b3     = (const float*)d_in[23];
  const int* src = ei;
  const int* dst = ei + NE;

  // avg_deg['log'] (host-computed, deterministic)
  static const double DEG[71] = {
    1, 72, 201, 816, 1790, 3756, 6923, 12768, 20286, 31710, 51623, 82296, 124280,
    177576, 251115, 326064, 395760, 456840, 506179, 516200, 507003, 493746, 489256,
    453936, 420025, 411320, 427761, 420700, 420500, 426780, 414284, 407008, 394053,
    360910, 322245, 313704, 282902, 270940, 237783, 209000, 193766, 177870, 162110,
    144848, 121230, 112700, 93483, 88512, 72275, 80700, 68799, 56784, 42665, 30996,
    25630, 12936, 9804, 8584, 5251, 3480, 3111, 2728, 1890, 1472, 1235, 330, 201,
    68, 69, 0, 71 };
  double num = 0.0, den = 0.0;
  for (int i = 0; i < 71; ++i) { num += log((double)i + 1.0) * DEG[i]; den += DEG[i]; }
  float avg_log = (float)(num / den);

  // workspace carve-up
  char* ws = (char*)d_ws;
  size_t off = 0;
  auto carve = [&](size_t bytes) -> char* {
    char* p = ws + off;
    off = (off + bytes + 255) & ~(size_t)255;
    return p;
  };
  _Float16* Af16  = (_Float16*)carve((size_t)NE * KPAD * 2);        // 51.2 MB
  _Float16* m16   = (_Float16*)carve((size_t)NE * MSTRIDE * 2);     // 76.8 MB
  _Float16* Bw    = (_Float16*)carve((size_t)NL * NTILES * 8 * 32 * 16 * 2);
  float* eaemb    = (float*)carve((size_t)NE * 50 * 4);             // 20 MB
  float* hbuf     = (float*)carve((size_t)NN * FT * 4);
  float* hpre     = (float*)carve((size_t)NN * FT * 4);
  float* postout  = (float*)carve((size_t)NN * FT * 4);
  float* cnt      = (float*)carve((size_t)NN * 4);
  int*   rowstart = (int*)carve((size_t)NN * 4);
  int*   fillb    = (int*)carve((size_t)NN * 4);
  int*   eidx     = (int*)carve((size_t)NE * 4);
  float* bn_mu    = (float*)carve(FT * 4);
  float* bn_rs    = (float*)carve(FT * 4);
  float* gpool    = (float*)carve((size_t)NG * FT * 4);
  (void)ws_size; (void)in_sizes; (void)n_in; (void)out_size;

  float* out = (float*)d_out;

  // embeddings, degrees, CSR, weight pre-shuffle (CSR depends only on dst)
  k_node_emb<<<cdiv(NN * FT, 256), 256, 0, stream>>>(x, node_emb_w, node_emb_b, hbuf);
  k_edge_emb<<<cdiv(NE * 50, 256), 256, 0, stream>>>(edge_attr, edge_emb_w, edge_emb_b, eaemb);
  k_fill<<<cdiv(NN, 256), 256, 0, stream>>>(cnt, 0.0f, NN);
  k_count<<<cdiv(NE, 256), 256, 0, stream>>>(dst, cnt);
  k_scan<<<1, 256, 0, stream>>>(cnt, rowstart);
  k_filli<<<cdiv(NN, 256), 256, 0, stream>>>(fillb, 0, NN);
  k_scatter<<<cdiv(NE, 256), 256, 0, stream>>>(dst, rowstart, fillb, eidx);
  k_bwconv<<<cdiv(NL * NTILES * 8 * 32, 256), 256, 0, stream>>>(pre_w, Bw);

  const int stripes = NE / 16;  // 6250 waves
  for (int l = 0; l < NL; ++l) {
    k_stageA<<<cdiv((long long)NE * (KPAD / 8), 256), 256, 0, stream>>>(
        hbuf, eaemb, ee_w + (size_t)l * 50 * FT, ee_b + (size_t)l * FT, src, dst, Af16);
    k_wmma_pre<<<cdiv(stripes * 32, 256), 256, 0, stream>>>(
        Af16, Bw + (size_t)l * NTILES * 8 * 32 * 16,
        pre_b + (size_t)l * TW * FT, m16);
    k_agg_post<<<NN, 160, 0, stream>>>(
        hbuf, m16, rowstart, cnt, eidx,
        post_w + (size_t)l * TW * (13 * FT) * FO, post_b + (size_t)l * TW * FO,
        avg_log, postout);
    k_lin<<<cdiv(NN * FT, 256), 256, 0, stream>>>(
        postout, lin_w + (size_t)l * FT * FT, lin_b + (size_t)l * FT, hpre);
    k_bnstats<<<FT, 256, 0, stream>>>(hpre, bn_mu, bn_rs);
    k_bnapply<<<cdiv(NN * FT, 256), 256, 0, stream>>>(
        hpre, bn_mu, bn_rs, bn_g + (size_t)l * FT, bn_b + (size_t)l * FT, hbuf);
  }

  // outputs: g[64] then per_atom[10000*75]
  k_copy<<<cdiv(NN * FT, 256), 256, 0, stream>>>(hbuf, out + NG);
  k_fill<<<cdiv(NG * FT, 256), 256, 0, stream>>>(gpool, 0.0f, NG * FT);
  k_pool<<<cdiv(NN * FT, 256), 256, 0, stream>>>(hbuf, batch, gpool);
  k_mlp<<<NG, 64, 0, stream>>>(gpool, mlp_w1, mlp_b1, mlp_w2, mlp_b2, mlp_w3, mlp_b3, out);
}
